// Attention_11158325035071
// MI455X (gfx1250) — matrix operand order repaired and checked by reading, not compiled, not run on hardware
//
#include <hip/hip_runtime.h>

#define DEV __device__ __forceinline__

typedef __attribute__((ext_vector_type(16))) __bf16 v16bf;
typedef __attribute__((ext_vector_type(8)))  __bf16 v8bf;
typedef __attribute__((ext_vector_type(4)))  __bf16 v4bf;
typedef __attribute__((ext_vector_type(8)))  float  v8f;

static constexpr int kB = 2, kS = 1024, kD = 2048;
static constexpr int kNH = 16, kNKV = 8, kHD = 128, kWIN = 256;

DEV __bf16 f2bf(float f) {
  unsigned u = __builtin_bit_cast(unsigned, f);
  u += 0x7fffu + ((u >> 16) & 1u);           // round-to-nearest-even
  unsigned short h = (unsigned short)(u >> 16);
  return __builtin_bit_cast(__bf16, h);
}
DEV float bf2f(__bf16 b) {
  unsigned u = ((unsigned)__builtin_bit_cast(unsigned short, b)) << 16;
  return __builtin_bit_cast(float, u);
}

DEV void load4(const float* p, float& a, float& b, float& c, float& d) {
  float4 v = *(const float4*)p; a = v.x; b = v.y; c = v.z; d = v.w;
}
DEV void load4(const __bf16* p, float& a, float& b, float& c, float& d) {
  v4bf v = *(const v4bf*)p;
  a = bf2f(v[0]); b = bf2f(v[1]); c = bf2f(v[2]); d = bf2f(v[3]);
}

DEV v8f wmma_bf16(v16bf a, v16bf b, v8f c) {
  return __builtin_amdgcn_wmma_f32_16x16x32_bf16(false, a, false, b, (short)0, c,
                                                 false, false);
}

// ---------------------------------------------------------------------------
// Tiled GEMM: C[M,N] = A[M,K] @ B[K,N], fp32 accumulate via
// v_wmma_f32_16x16x32_bf16. A fp32 or bf16; B (weights) fp32.
// Block tile 128(M) x 64(N) x 64(K); 8 waves each computing 32x32.
// Software-pipelined: next tile's global loads issued into registers while
// WMMAs consume the LDS-staged current tile. B staged transposed so the
// B fragment is two 16B LDS reads of contiguous k.
// ---------------------------------------------------------------------------
template <typename TA>
__global__ __launch_bounds__(256)
void gemm_bf16_wmma(const TA* __restrict__ A, const float* __restrict__ Bw,
                    float* __restrict__ C, int M, int N, int K) {
  __shared__ __align__(16) __bf16 As[128][80];    // row stride 160B (16B mult)
  __shared__ __align__(16) __bf16 BsT[64][80];    // transposed: [n][k]
  const int tid = threadIdx.x;
  const int m0 = blockIdx.y * 128;
  const int n0 = blockIdx.x * 64;
  const int wave = __builtin_amdgcn_readfirstlane(tid >> 5);  // SGPR wave id
  const int lane = tid & 31;
  const int wm = (wave & 3) * 32, wn = (wave >> 2) * 32;
  const int ln = lane & 15, hi = lane >> 4;
  const int lr = tid >> 4;            // 0..15 base row
  const int lc = (tid & 15) * 4;      // 0..60 col quad

  float4 Ar[8], Br[4];

  auto loadA = [&](int k0) {
#pragma unroll
    for (int i = 0; i < 8; ++i) {
      int r = lr + 16 * i;            // 0..127
      load4(A + (size_t)(m0 + r) * K + k0 + lc, Ar[i].x, Ar[i].y, Ar[i].z, Ar[i].w);
    }
  };
  auto loadB = [&](int k0) {
#pragma unroll
    for (int i = 0; i < 4; ++i) {
      int r = lr + 16 * i;            // 0..63
      load4(Bw + (size_t)(k0 + r) * N + n0 + lc, Br[i].x, Br[i].y, Br[i].z, Br[i].w);
    }
  };
  auto stage = [&]() {
#pragma unroll
    for (int i = 0; i < 8; ++i) {
      int r = lr + 16 * i;
      v4bf p;
      p[0] = f2bf(Ar[i].x); p[1] = f2bf(Ar[i].y);
      p[2] = f2bf(Ar[i].z); p[3] = f2bf(Ar[i].w);
      *(v4bf*)&As[r][lc] = p;         // ds_store_b64
    }
#pragma unroll
    for (int i = 0; i < 4; ++i) {
      int r = lr + 16 * i;
      BsT[lc + 0][r] = f2bf(Br[i].x);
      BsT[lc + 1][r] = f2bf(Br[i].y);
      BsT[lc + 2][r] = f2bf(Br[i].z);
      BsT[lc + 3][r] = f2bf(Br[i].w);
    }
  };

  v8f acc[2][2] = {};
  loadA(0);
  loadB(0);

  for (int k0 = 0; k0 < K; k0 += 64) {
    stage();
    __syncthreads();
    const int kn = k0 + 64;
    if (kn < K) {
      __builtin_prefetch(Bw + (size_t)kn * N + n0, 0, 1);  // global_prefetch_b8
      loadA(kn);                       // in flight during WMMAs below
      loadB(kn);
    }
#pragma unroll
    for (int kc = 0; kc < 2; ++kc) {
      v16bf aF[2], bF[2];
#pragma unroll
      for (int mi = 0; mi < 2; ++mi) {
        const __bf16* p = &As[wm + mi * 16 + ln][kc * 32 + 8 * hi];
        v8bf lo = *(const v8bf*)p;
        v8bf hh = *(const v8bf*)(p + 16);
#pragma unroll
        for (int e = 0; e < 8; ++e) { aF[mi][e] = lo[e]; aF[mi][8 + e] = hh[e]; }
      }
#pragma unroll
      for (int ni = 0; ni < 2; ++ni) {
        const __bf16* q = &BsT[wn + ni * 16 + ln][kc * 32 + 8 * hi];
        v8bf lo = *(const v8bf*)q;
        v8bf hh = *(const v8bf*)(q + 16);
        for (int e = 0; e < 8; ++e) { bF[ni][e] = lo[e]; bF[ni][8 + e] = hh[e]; }
      }
#pragma unroll
      for (int mi = 0; mi < 2; ++mi)
#pragma unroll
        for (int ni = 0; ni < 2; ++ni)
          acc[mi][ni] = wmma_bf16(aF[mi], bF[ni], acc[mi][ni]);
    }
    __syncthreads();
  }
#pragma unroll
  for (int mi = 0; mi < 2; ++mi)
#pragma unroll
    for (int ni = 0; ni < 2; ++ni)
#pragma unroll
      for (int r = 0; r < 8; ++r)
        C[(size_t)(m0 + wm + mi * 16 + r + 8 * hi) * N + n0 + wn + ni * 16 + ln] =
            acc[mi][ni][r];
}

// ---------------------------------------------------------------------------
// RoPE + pack: fp32 q/k/v -> bf16 q1 (rope@pos), q2 (rope@WIN), k1 (rope@pos),
// k2 (raw), vT transposed [b][kv][d][s] for contiguous-K B-operand loads.
// ---------------------------------------------------------------------------
__global__ __launch_bounds__(256)
void rope_pack(const float* __restrict__ q, const float* __restrict__ k,
               const float* __restrict__ v, const float* __restrict__ fcos,
               const float* __restrict__ fsin,
               __bf16* __restrict__ q1, __bf16* __restrict__ q2,
               __bf16* __restrict__ k1, __bf16* __restrict__ k2,
               __bf16* __restrict__ vT) {
  const int bs = blockIdx.x;
  const int b = bs >> 10, s = bs & (kS - 1);
  const size_t qbase = (size_t)bs * (kNH * kHD);
  const size_t kvbase = (size_t)bs * (kNKV * kHD);
  for (int idx = threadIdx.x; idx < kNH * (kHD / 2); idx += 256) {
    int h = idx >> 6, p = idx & 63;
    size_t o = qbase + h * kHD + 2 * p;
    float xr = q[o], xi = q[o + 1];
    float c = fcos[s * 64 + p], sn = fsin[s * 64 + p];
    q1[o] = f2bf(xr * c - xi * sn);
    q1[o + 1] = f2bf(xr * sn + xi * c);
    float cw = fcos[kWIN * 64 + p], sw = fsin[kWIN * 64 + p];
    q2[o] = f2bf(xr * cw - xi * sw);
    q2[o + 1] = f2bf(xr * sw + xi * cw);
  }
  for (int idx = threadIdx.x; idx < kNKV * (kHD / 2); idx += 256) {
    int h = idx >> 6, p = idx & 63;
    size_t o = kvbase + h * kHD + 2 * p;
    float xr = k[o], xi = k[o + 1];
    float c = fcos[s * 64 + p], sn = fsin[s * 64 + p];
    k1[o] = f2bf(xr * c - xi * sn);
    k1[o + 1] = f2bf(xr * sn + xi * c);
    k2[o] = f2bf(xr);
    k2[o + 1] = f2bf(xi);
    float v0 = v[o], v1 = v[o + 1];
    vT[(((size_t)b * kNKV + h) * kHD + 2 * p) * kS + s] = f2bf(v0);
    vT[(((size_t)b * kNKV + h) * kHD + 2 * p + 1) * kS + s] = f2bf(v1);
  }
}

// ---------------------------------------------------------------------------
// Flash attention with dual-score select. One wave = 16 query rows; 32 keys
// per iteration. Score select: j>i -> -inf; i-WIN<j<=i -> s1; else s2.
// s1/s2 WMMA passes gated by SGPR (readfirstlane) conditions so the gates
// lower to scalar s_cbranch and EXEC stays all-ones around every WMMA.
// ---------------------------------------------------------------------------
__global__ __launch_bounds__(128)
void attn_fused(const __bf16* __restrict__ q1g, const __bf16* __restrict__ q2g,
                const __bf16* __restrict__ k1g, const __bf16* __restrict__ k2g,
                const __bf16* __restrict__ vT, __bf16* __restrict__ og) {
  __shared__ __align__(16) __bf16 pbuf[4][16][40];
  const int h = blockIdx.y, b = blockIdx.z;
  const int kvh = h >> 1;                        // NREP = 2
  const int wave = __builtin_amdgcn_readfirstlane(threadIdx.x >> 5);  // SGPR
  const int lane = threadIdx.x & 31;
  const int q0 = blockIdx.x * 64 + wave * 16;    // SGPR-uniform
  const int ln = lane & 15, hi = lane >> 4;

  // Q fragments (4 K-chunks of 32 over HD=128), for both rope variants.
  v16bf q1A[4], q2A[4];
  {
    const __bf16* r1 = q1g + (((size_t)b * kS + q0 + ln) * kNH + h) * kHD;
    const __bf16* r2 = q2g + (((size_t)b * kS + q0 + ln) * kNH + h) * kHD;
#pragma unroll
    for (int c = 0; c < 4; ++c) {
      int base = c * 32 + 8 * hi;
      v8bf lo1 = *(const v8bf*)(r1 + base);
      v8bf hh1 = *(const v8bf*)(r1 + base + 16);
      v8bf lo2 = *(const v8bf*)(r2 + base);
      v8bf hh2 = *(const v8bf*)(r2 + base + 16);
#pragma unroll
      for (int e = 0; e < 8; ++e) {
        q1A[c][e] = lo1[e]; q1A[c][8 + e] = hh1[e];
        q2A[c][e] = lo2[e]; q2A[c][8 + e] = hh2[e];
      }
    }
  }

  v8f o[8];
#pragma unroll
  for (int d = 0; d < 8; ++d) o[d] = {};
  float mrow[8], lrow[8];
#pragma unroll
  for (int r = 0; r < 8; ++r) { mrow[r] = -1e30f; lrow[r] = 0.f; }

  const float scale = 0.08838834764831845f;      // 1/sqrt(128)
  const int iMax = q0 + 15;                      // SGPR-uniform

  for (int k0 = 0; k0 <= iMax; k0 += 32) {
    // SGPR-uniform gates: does this 32-key tile intersect the in-window
    // region (q0-WIN, q0+15] (needs s1) and/or far region [0, q0+15-WIN]?
    const bool needS1 = (k0 + 31 > q0 - kWIN);
    const bool needS2 = (k0 <= iMax - kWIN);

    v8f s1[2], s2[2];
    s1[0] = {}; s1[1] = {}; s2[0] = {}; s2[1] = {};
    if (needS1) {
#pragma unroll
      for (int c = 0; c < 4; ++c)
#pragma unroll
        for (int ni = 0; ni < 2; ++ni) {
          const size_t krow = ((size_t)b * kS + k0 + ni * 16 + ln) * kNKV + kvh;
          const __bf16* kp = k1g + krow * kHD + c * 32 + 8 * hi;
          v8bf klo = *(const v8bf*)kp, khh = *(const v8bf*)(kp + 16);
          v16bf bk;
          for (int e = 0; e < 8; ++e) { bk[e] = klo[e]; bk[8 + e] = khh[e]; }
          s1[ni] = wmma_bf16(q1A[c], bk, s1[ni]);
        }
    }
    if (needS2) {
#pragma unroll
      for (int c = 0; c < 4; ++c)
#pragma unroll
        for (int ni = 0; ni < 2; ++ni) {
          const size_t krow = ((size_t)b * kS + k0 + ni * 16 + ln) * kNKV + kvh;
          const __bf16* kp = k2g + krow * kHD + c * 32 + 8 * hi;
          v8bf klo = *(const v8bf*)kp, khh = *(const v8bf*)(kp + 16);
          v16bf bk;
          for (int e = 0; e < 8; ++e) { bk[e] = klo[e]; bk[8 + e] = khh[e]; }
          s2[ni] = wmma_bf16(q2A[c], bk, s2[ni]);
        }
    }

    // Select + mask; track tile row-max.
    float pv[2][8], tmax[8];
#pragma unroll
    for (int r = 0; r < 8; ++r) tmax[r] = -1e30f;
#pragma unroll
    for (int ni = 0; ni < 2; ++ni) {
      const int j = k0 + ni * 16 + ln;
#pragma unroll
      for (int r = 0; r < 8; ++r) {
        const int i = q0 + r + 8 * hi;
        float val = (j > i) ? -1e30f
                  : (j > i - kWIN) ? s1[ni][r] * scale
                                   : s2[ni][r] * scale;
        pv[ni][r] = val;
        tmax[r] = fmaxf(tmax[r], val);
      }
    }
    // Row reductions across the 16-lane half (rows live in one half-wave).
#pragma unroll
    for (int off = 1; off < 16; off <<= 1)
#pragma unroll
      for (int r = 0; r < 8; ++r)
        tmax[r] = fmaxf(tmax[r], __shfl_xor(tmax[r], off, 32));
    float alpha[8];
#pragma unroll
    for (int r = 0; r < 8; ++r) {
      float mnew = fmaxf(mrow[r], tmax[r]);
      alpha[r] = __expf(mrow[r] - mnew);
      mrow[r] = mnew;
    }
    float rsum[8];
#pragma unroll
    for (int r = 0; r < 8; ++r) rsum[r] = 0.f;
#pragma unroll
    for (int ni = 0; ni < 2; ++ni)
#pragma unroll
      for (int r = 0; r < 8; ++r) {
        float p = __expf(pv[ni][r] - mrow[r]);
        pv[ni][r] = p;
        rsum[r] += p;
      }
#pragma unroll
    for (int off = 1; off < 16; off <<= 1)
#pragma unroll
      for (int r = 0; r < 8; ++r) rsum[r] += __shfl_xor(rsum[r], off, 32);
#pragma unroll
    for (int r = 0; r < 8; ++r) lrow[r] = lrow[r] * alpha[r] + rsum[r];
#pragma unroll
    for (int d = 0; d < 8; ++d)
#pragma unroll
      for (int r = 0; r < 8; ++r) o[d][r] *= alpha[r];

    // Transpose P (C-frag layout -> A-frag layout) through per-wave LDS.
#pragma unroll
    for (int ni = 0; ni < 2; ++ni)
#pragma unroll
      for (int r = 0; r < 8; ++r)
        pbuf[wave][r + 8 * hi][ni * 16 + ln] = f2bf(pv[ni][r]);
    __asm__ volatile("s_wait_dscnt 0" ::: "memory");  // per-wave LDS fence
    v16bf pA;
    {
      const __bf16* pp = &pbuf[wave][ln][8 * hi];
      v8bf lo = *(const v8bf*)pp;
      v8bf hh = *(const v8bf*)(pp + 16);
#pragma unroll
      for (int e = 0; e < 8; ++e) { pA[e] = lo[e]; pA[8 + e] = hh[e]; }
    }
    // O += P @ V  (vT gives contiguous key-dim loads for the B fragment).
#pragma unroll
    for (int d = 0; d < 8; ++d) {
      const __bf16* vp =
          vT + (((size_t)b * kNKV + kvh) * kHD + d * 16 + ln) * kS + k0 + 8 * hi;
      v8bf vlo = *(const v8bf*)vp, vhh = *(const v8bf*)(vp + 16);
      v16bf bv;
      for (int e = 0; e < 8; ++e) { bv[e] = vlo[e]; bv[8 + e] = vhh[e]; }
      o[d] = wmma_bf16(pA, bv, o[d]);
    }
  }

  float inv[8];
#pragma unroll
  for (int r = 0; r < 8; ++r) inv[r] = 1.f / lrow[r];
#pragma unroll
  for (int d = 0; d < 8; ++d)
#pragma unroll
    for (int r = 0; r < 8; ++r)
      og[(((size_t)b * kS + q0 + r + 8 * hi) * kNH + h) * kHD + d * 16 + ln] =
          f2bf(o[d][r] * inv[r]);
}

// ---------------------------------------------------------------------------
extern "C" void kernel_launch(void* const* d_in, const int* in_sizes, int n_in,
                              void* d_out, int out_size, void* d_ws, size_t ws_size,
                              hipStream_t stream) {
  (void)in_sizes; (void)n_in; (void)out_size; (void)ws_size;
  const float* x    = (const float*)d_in[0];
  const float* fcos = (const float*)d_in[1];
  const float* fsin = (const float*)d_in[2];
  const float* wq   = (const float*)d_in[3];
  const float* wk   = (const float*)d_in[4];
  const float* wv   = (const float*)d_in[5];
  const float* wo   = (const float*)d_in[6];
  float* out = (float*)d_out;

  char* w = (char*)d_ws;
  const size_t MB = 1ull << 20;
  float*  qf = (float*)(w + 0);          // 16 MB: [B,S,NH,HD] fp32
  float*  kf = (float*)(w + 16 * MB);    //  8 MB
  float*  vf = (float*)(w + 24 * MB);    //  8 MB
  __bf16* q1 = (__bf16*)(w + 32 * MB);   //  8 MB
  __bf16* q2 = (__bf16*)(w + 40 * MB);   //  8 MB
  __bf16* k1 = (__bf16*)(w + 48 * MB);   //  4 MB
  __bf16* k2 = (__bf16*)(w + 52 * MB);   //  4 MB
  __bf16* vT = (__bf16*)(w + 56 * MB);   //  4 MB
  __bf16* ao = (__bf16*)(w + 60 * MB);   //  8 MB

  const int M = kB * kS;  // 2048
  gemm_bf16_wmma<float><<<dim3((kNH * kHD) / 64, M / 128), 256, 0, stream>>>(
      x, wq, qf, M, kNH * kHD, kD);
  gemm_bf16_wmma<float><<<dim3((kNKV * kHD) / 64, M / 128), 256, 0, stream>>>(
      x, wk, kf, M, kNKV * kHD, kD);
  gemm_bf16_wmma<float><<<dim3((kNKV * kHD) / 64, M / 128), 256, 0, stream>>>(
      x, wv, vf, M, kNKV * kHD, kD);
  rope_pack<<<dim3(kB * kS), 256, 0, stream>>>(qf, kf, vf, fcos, fsin,
                                               q1, q2, k1, k2, vT);
  attn_fused<<<dim3(kS / 64, kNH, kB), 128, 0, stream>>>(q1, q2, k1, k2, vT, ao);
  gemm_bf16_wmma<__bf16><<<dim3(kD / 64, M / 128), 256, 0, stream>>>(
      ao, wo, out, M, kD, kNH * kHD);
}
